// SuperGATs_48593259987030
// MI455X (gfx1250) — matrix operand order repaired and checked by reading, not compiled, not run on hardware
//
#include <hip/hip_runtime.h>
#include <hip/hip_bf16.h>

// ---------------- problem constants (from reference) ----------------
#define NNODES 20000
#define FIN    128
#define HID    32
#define HEADS  8
#define HC     (HEADS * HID)       // 256
#define NEDGE  320000
#define ETOT   (NEDGE + NNODES)    // 340000 (edges + self loops)
#define NGRAPH 64
#define NCLS   10
#define MBLK   32                  // GEMM block rows (2 row-groups of 16)

typedef __bf16 bf16_t;
typedef __attribute__((ext_vector_type(16))) __bf16 v16bf;
typedef __attribute__((ext_vector_type(8)))  float  v8f;

// ------------- ordered-uint encoding for float atomic max -------------
__device__ __forceinline__ unsigned fenc(float f) {
    unsigned u = __float_as_uint(f);
    return (u & 0x80000000u) ? ~u : (u | 0x80000000u);
}
__device__ __forceinline__ float fdec(unsigned k) {
    unsigned u = (k & 0x80000000u) ? (k & 0x7FFFFFFFu) : ~k;
    return __uint_as_float(u);
}

// ---------------- utility kernels ----------------
__global__ void k_cast_bf16(const float* __restrict__ s, bf16_t* __restrict__ d, long n) {
    long i = (long)blockIdx.x * blockDim.x + threadIdx.x;
    if (i < n) d[i] = (bf16_t)s[i];
}
__global__ void k_fill_f32(float* __restrict__ p, float v, long n) {
    long i = (long)blockIdx.x * blockDim.x + threadIdx.x;
    if (i < n) p[i] = v;
}
__global__ void k_fill_amax(unsigned* __restrict__ p, long n) {
    long i = (long)blockIdx.x * blockDim.x + threadIdx.x;
    if (i < n) p[i] = fenc(-3.0e38f);
}

// ---------------- WMMA bf16 GEMM: C[M,N] = A[M,K] * B[K,N] ----------------
// A,B row-major bf16, C row-major f32.  N must be 256, M%32==0, K%32==0.
// 256 threads = 8 waves. Block tile: 32 rows x 256 cols.
//   wave w: row-group rg = w/4 (16 rows), col-group (w%4)*64 -> 4 accumulators
// Per K-step: A tile (32x32 bf16, 2KB) staged into LDS with
// global_load_async_to_lds_b128 (ASYNCcnt), then 4 WMMAs per wave reusing
// one A fragment read back from LDS.
__global__ __launch_bounds__(256) void k_gemm_wmma_bf16(
    const bf16_t* __restrict__ A, const bf16_t* __restrict__ B,
    float* __restrict__ C, int M, int N, int K)
{
    __shared__ bf16_t As[MBLK * 32];           // 32 rows x 32 K, row-major

    const int tid  = threadIdx.x;
    const int lane = tid & 31;
    const int wave = tid >> 5;
    const int rg   = wave >> 2;                // row group 0..1
    const int cb0  = (wave & 3) * 64;          // first column of this wave's 4 tiles
    const int m0   = blockIdx.x * MBLK;

    // low 32 bits of a flat LDS pointer == LDS byte address (aperture rule)
    const unsigned lds_base = (unsigned)(size_t)(void*)As;

    v8f acc0 = {}, acc1 = {}, acc2 = {}, acc3 = {};

    const int arow = rg * 16 + (lane & 15);    // A row within block tile
    const int kb   = (lane >> 4) * 8;          // A K sub-block per lane half

    for (int k0 = 0; k0 < K; k0 += 32) {
        // ---- async stage A tile into LDS: 128 threads x 16B = 2KB ----
        if (tid < 128) {                        // waves 0..3 (wave-uniform)
            int row = tid >> 2;                 // 4 x 16B per 64B row
            int kk  = (tid & 3) * 8;
            unsigned long long ga =
                (unsigned long long)(A + (size_t)(m0 + row) * K + k0 + kk);
            unsigned la = lds_base + tid * 16;
            asm volatile("global_load_async_to_lds_b128 %0, %1, off"
                         :: "v"(la), "v"(ga) : "memory");
        }
        asm volatile("s_wait_asynccnt 0x0" ::: "memory");
        __syncthreads();

        // ---- A fragment from LDS (two 16B ds loads per lane) ----
        const bf16_t* ap = As + arow * 32 + kb;
        v16bf a;
        #pragma unroll
        for (int i = 0; i < 8; ++i) { a[i] = ap[i]; a[8 + i] = ap[16 + i]; }

        // ---- B fragments from global (L2/WGP$-resident, shared by all blocks) ----
        const bf16_t* bp = B + (size_t)(k0 + (lane >> 4) * 8) * N + cb0 + (lane & 15);
        v16bf b0, b1, b2, b3;
        #pragma unroll
        for (int i = 0; i < 16; ++i) {
            const size_t kr = (size_t)(i < 8 ? i : i + 8) * N;
            b0[i] = bp[kr]; b1[i] = bp[kr + 16]; b2[i] = bp[kr + 32]; b3[i] = bp[kr + 48];
        }

        acc0 = __builtin_amdgcn_wmma_f32_16x16x32_bf16(false, a, false, b0, (short)0, acc0, false, false);
        acc1 = __builtin_amdgcn_wmma_f32_16x16x32_bf16(false, a, false, b1, (short)0, acc1, false, false);
        acc2 = __builtin_amdgcn_wmma_f32_16x16x32_bf16(false, a, false, b2, (short)0, acc2, false, false);
        acc3 = __builtin_amdgcn_wmma_f32_16x16x32_bf16(false, a, false, b3, (short)0, acc3, false, false);

        __syncthreads();                        // protect LDS before next stage
    }

    const int mrow = m0 + rg * 16 + ((lane >> 4) << 3);
    const int nc   = cb0 + (lane & 15);
    #pragma unroll
    for (int v = 0; v < 8; ++v) {
        float* cr = C + (size_t)(mrow + v) * N + nc;
        cr[0] = acc0[v]; cr[16] = acc1[v]; cr[32] = acc2[v]; cr[48] = acc3[v];
    }
}

// ---------------- edge kernels (attention) ----------------
__device__ __forceinline__ void edge_endpoints(int e, const int* __restrict__ ei,
                                               int& src, int& dst) {
    if (e < NEDGE) { src = ei[e]; dst = ei[NEDGE + e]; }
    else           { src = e - NEDGE; dst = e - NEDGE; }   // self loops
}

// alpha = leaky_relu( (x_j.att_l + x_i.att_r) * sigmoid(x_i.x_j), 0.2 ); amax over dst
__global__ void k_edge_alpha(const float* __restrict__ h, const int* __restrict__ ei,
                             const float* __restrict__ att_l, const float* __restrict__ att_r,
                             float* __restrict__ alpha, unsigned* __restrict__ amax)
{
    long t = (long)blockIdx.x * blockDim.x + threadIdx.x;
    if (t >= (long)ETOT * HEADS) return;
    int e = (int)(t / HEADS), hh = (int)(t % HEADS);
    int src, dst; edge_endpoints(e, ei, src, dst);
    const float* xj = h + (size_t)src * HC + hh * HID;
    const float* xi = h + (size_t)dst * HC + hh * HID;
    const float* al = att_l + hh * HID;
    const float* ar = att_r + hh * HID;
    float dot = 0.f, a = 0.f;
    #pragma unroll 8
    for (int c = 0; c < HID; ++c) {
        float vj = xj[c], vi = xi[c];
        dot += vi * vj;
        a   += vj * al[c] + vi * ar[c];
    }
    a = a * (1.f / (1.f + __expf(-dot)));          // * sigmoid(logits)
    a = (a > 0.f) ? a : 0.2f * a;                  // leaky_relu 0.2
    alpha[t] = a;
    atomicMax(&amax[(size_t)dst * HEADS + hh], fenc(a));
}

// ex = exp(alpha - amax[dst]); denom[dst] += ex   (ex stored in-place in alpha)
__global__ void k_edge_exp(const int* __restrict__ ei, float* __restrict__ alpha,
                           const unsigned* __restrict__ amax, float* __restrict__ denom)
{
    long t = (long)blockIdx.x * blockDim.x + threadIdx.x;
    if (t >= (long)ETOT * HEADS) return;
    int e = (int)(t / HEADS), hh = (int)(t % HEADS);
    int src, dst; edge_endpoints(e, ei, src, dst); (void)src;
    float m  = fdec(amax[(size_t)dst * HEADS + hh]);
    float ex = __expf(alpha[t] - m);
    alpha[t] = ex;
    atomicAdd(&denom[(size_t)dst * HEADS + hh], ex);
}

// out[dst] += h[src] * ex/(denom[dst]+1e-16), per (edge, channel)
__global__ void k_edge_aggregate(const float* __restrict__ h, const int* __restrict__ ei,
                                 const float* __restrict__ exv, const float* __restrict__ denom,
                                 float* __restrict__ out)
{
    long t = (long)blockIdx.x * blockDim.x + threadIdx.x;
    if (t >= (long)ETOT * HC) return;
    int e = (int)(t / HC), c = (int)(t % HC);
    int hh = c / HID;
    int src, dst; edge_endpoints(e, ei, src, dst);
    float w = exv[(size_t)e * HEADS + hh] / (denom[(size_t)dst * HEADS + hh] + 1e-16f);
    atomicAdd(&out[(size_t)dst * HC + c], h[(size_t)src * HC + c] * w);
}

// h1 = elu(agg + b); also emit bf16 copy for next WMMA GEMM
__global__ void k_bias_elu_cast(const float* __restrict__ agg, const float* __restrict__ b,
                                bf16_t* __restrict__ hb, long n)
{
    long i = (long)blockIdx.x * blockDim.x + threadIdx.x;
    if (i >= n) return;
    float v = agg[i] + b[i % HC];
    v = (v > 0.f) ? v : (__expf(v) - 1.f);         // elu
    hb[i] = (bf16_t)v;
}

// ---------------- pooling + classifier ----------------
__global__ void k_pool(const float* __restrict__ agg, const float* __restrict__ b2,
                       const int* __restrict__ batch, float* __restrict__ pooled,
                       float* __restrict__ cnt)
{
    long t = (long)blockIdx.x * blockDim.x + threadIdx.x;
    if (t >= (long)NNODES * HC) return;
    int n = (int)(t / HC), c = (int)(t % HC);
    int g = batch[n];
    atomicAdd(&pooled[(size_t)g * HC + c], agg[t] + b2[c]);
    if (c == 0) atomicAdd(&cnt[g], 1.f);
}

__global__ void k_classifier(const float* __restrict__ pooled, const float* __restrict__ cnt,
                             const float* __restrict__ W, const float* __restrict__ b,
                             float* __restrict__ out)
{
    int t = blockIdx.x * blockDim.x + threadIdx.x;
    if (t >= NGRAPH * NCLS) return;
    int g = t / NCLS, k = t % NCLS;
    float inv = 1.f / fmaxf(cnt[g], 1.f);
    float acc = b[k];
    #pragma unroll 8
    for (int c = 0; c < HC; ++c)
        acc += pooled[(size_t)g * HC + c] * inv * W[(size_t)c * NCLS + k];
    out[t] = acc;
}

// ---------------- launcher ----------------
extern "C" void kernel_launch(void* const* d_in, const int* in_sizes, int n_in,
                              void* d_out, int out_size, void* d_ws, size_t ws_size,
                              hipStream_t stream)
{
    const float* x      = (const float*)d_in[0];
    const int*   ei     = (const int*)  d_in[1];   // [2, E] flattened
    const int*   batch  = (const int*)  d_in[2];
    const float* W1     = (const float*)d_in[3];
    const float* attl1  = (const float*)d_in[4];
    const float* attr1  = (const float*)d_in[5];
    const float* b1     = (const float*)d_in[6];
    const float* W2     = (const float*)d_in[7];
    const float* attl2  = (const float*)d_in[8];
    const float* attr2  = (const float*)d_in[9];
    const float* b2     = (const float*)d_in[10];
    const float* linW   = (const float*)d_in[11];
    const float* linb   = (const float*)d_in[12];
    float* out = (float*)d_out;

    // workspace layout (256B aligned chunks)
    char* ws = (char*)d_ws;
    size_t o = 0;
    auto take = [&](size_t bytes) { size_t r = o; o += (bytes + 255) & ~(size_t)255; return r; };
    bf16_t*   xb    = (bf16_t*)  (ws + take((size_t)NNODES * FIN * 2));
    bf16_t*   W1b   = (bf16_t*)  (ws + take((size_t)FIN * HC * 2));
    bf16_t*   W2b   = (bf16_t*)  (ws + take((size_t)HC * HC * 2));
    float*    hlin  = (float*)   (ws + take((size_t)NNODES * HC * 4));  // GEMM output
    float*    hagg  = (float*)   (ws + take((size_t)NNODES * HC * 4));  // aggregation output
    bf16_t*   hb    = (bf16_t*)  (ws + take((size_t)NNODES * HC * 2));  // layer-2 GEMM input
    float*    alpha = (float*)   (ws + take((size_t)ETOT * HEADS * 4));
    unsigned* amax  = (unsigned*)(ws + take((size_t)NNODES * HEADS * 4));
    float*    denom = (float*)   (ws + take((size_t)NNODES * HEADS * 4));
    float*    pooled= (float*)   (ws + take((size_t)NGRAPH * HC * 4));
    float*    cnt   = (float*)   (ws + take((size_t)NGRAPH * 4));

    const int T = 256;
    auto nb = [](long n, int t) { return (unsigned)((n + t - 1) / t); };

    // --- bf16 conversions ---
    k_cast_bf16<<<nb((long)NNODES * FIN, T), T, 0, stream>>>(x,  xb,  (long)NNODES * FIN);
    k_cast_bf16<<<nb((long)FIN * HC,    T), T, 0, stream>>>(W1, W1b, (long)FIN * HC);
    k_cast_bf16<<<nb((long)HC * HC,     T), T, 0, stream>>>(W2, W2b, (long)HC * HC);

    // ================= layer 1 =================
    k_gemm_wmma_bf16<<<NNODES / MBLK, 256, 0, stream>>>(xb, W1b, hlin, NNODES, HC, FIN);

    k_fill_amax<<<nb((long)NNODES * HEADS, T), T, 0, stream>>>(amax, (long)NNODES * HEADS);
    k_fill_f32 <<<nb((long)NNODES * HEADS, T), T, 0, stream>>>(denom, 0.f, (long)NNODES * HEADS);
    k_fill_f32 <<<nb((long)NNODES * HC,    T), T, 0, stream>>>(hagg,  0.f, (long)NNODES * HC);

    k_edge_alpha    <<<nb((long)ETOT * HEADS, T), T, 0, stream>>>(hlin, ei, attl1, attr1, alpha, amax);
    k_edge_exp      <<<nb((long)ETOT * HEADS, T), T, 0, stream>>>(ei, alpha, amax, denom);
    k_edge_aggregate<<<nb((long)ETOT * HC,    T), T, 0, stream>>>(hlin, ei, alpha, denom, hagg);

    k_bias_elu_cast<<<nb((long)NNODES * HC, T), T, 0, stream>>>(hagg, b1, hb, (long)NNODES * HC);

    // ================= layer 2 =================
    k_gemm_wmma_bf16<<<NNODES / MBLK, 256, 0, stream>>>(hb, W2b, hlin, NNODES, HC, HC);

    k_fill_amax<<<nb((long)NNODES * HEADS, T), T, 0, stream>>>(amax, (long)NNODES * HEADS);
    k_fill_f32 <<<nb((long)NNODES * HEADS, T), T, 0, stream>>>(denom, 0.f, (long)NNODES * HEADS);
    k_fill_f32 <<<nb((long)NNODES * HC,    T), T, 0, stream>>>(hagg,  0.f, (long)NNODES * HC);

    k_edge_alpha    <<<nb((long)ETOT * HEADS, T), T, 0, stream>>>(hlin, ei, attl2, attr2, alpha, amax);
    k_edge_exp      <<<nb((long)ETOT * HEADS, T), T, 0, stream>>>(ei, alpha, amax, denom);
    k_edge_aggregate<<<nb((long)ETOT * HC,    T), T, 0, stream>>>(hlin, ei, alpha, denom, hagg);

    // ================= pool + classify =================
    k_fill_f32<<<nb((long)NGRAPH * HC, T), T, 0, stream>>>(pooled, 0.f, (long)NGRAPH * HC);
    k_fill_f32<<<nb((long)NGRAPH,      T), T, 0, stream>>>(cnt,    0.f, (long)NGRAPH);
    k_pool<<<nb((long)NNODES * HC, T), T, 0, stream>>>(hagg, b2, batch, pooled, cnt);
    k_classifier<<<nb((long)NGRAPH * NCLS, T), T, 0, stream>>>(pooled, cnt, linW, linb, out);
}
